// PerceiverAttention_45406394254124
// MI455X (gfx1250) — matrix operand order repaired and checked by reading, not compiled, not run on hardware
//
#include <hip/hip_runtime.h>

// ---------------------------------------------------------------------------
// Perceiver attention for MI455X (gfx1250, wave32, WMMA).
// Pipeline: layernorm->f16, WMMA projection GEMMs, rotary, flash attention
// (WMMA QK^T + PV with online softmax), WMMA output GEMM -> f32.
// Round 3: TDM tensor_load_to_lds for GEMM A-tiles (TENSORcnt path, with
// TDM LDS padding), async global->LDS b128 kept for attention K staging
// (ASYNCcnt path), fragment preloads to pipeline DS latency.
// ---------------------------------------------------------------------------

typedef __attribute__((ext_vector_type(16))) _Float16 v16h;
typedef __attribute__((ext_vector_type(8)))  _Float16 v8h;
typedef __attribute__((ext_vector_type(8)))  float    v8f;
typedef __attribute__((ext_vector_type(4)))  float    v4f;
typedef __attribute__((ext_vector_type(4)))  unsigned tdm_u4;
typedef __attribute__((ext_vector_type(8)))  int      tdm_i8;
typedef __attribute__((ext_vector_type(4)))  int      tdm_i4;

#define B_     2
#define N_     4096
#define M_     512
#define DIM_   1024
#define H_     16
#define HD_    64
#define NM_    4608      /* N_ + M_ */
#define INNER_ 1024

// ---- CDNA5 WMMA fragment builders (per ISA 7.12.2 VGPR layouts) ------------
// A (16x32 f16): lane L holds row L&15; halves[0..7]=K(8*hi..), halves[8..15]=K(16+8*hi..)
__device__ __forceinline__ v16h loadFragA(const _Float16* rowBase, int hi) {
  v8h lo = *(const v8h*)(rowBase + 8 * hi);
  v8h up = *(const v8h*)(rowBase + 16 + 8 * hi);
  v16h r;
#pragma unroll
  for (int i = 0; i < 8; ++i) { r[i] = lo[i]; r[i + 8] = up[i]; }
  return r;
}
// B (32x16 f16): lane L holds column L&15; halves[0..7]=K(8*hi..), halves[8..15]=K(16+8*hi..)
__device__ __forceinline__ v16h loadFragB(const _Float16* colBase, int hi) {
  v8h lo = *(const v8h*)(colBase + 8 * hi);
  v8h up = *(const v8h*)(colBase + 16 + 8 * hi);
  v16h r;
#pragma unroll
  for (int i = 0; i < 8; ++i) { r[i] = lo[i]; r[i + 8] = up[i]; }
  return r;
}

__device__ __forceinline__ v8f wmma_f32(v16h a, v16h b, v8f c) {
  // 8 args: (neg_a, A, neg_b, B, c_mod, C, reuse_a, reuse_b)
  return __builtin_amdgcn_wmma_f32_16x16x32_f16(false, a, false, b, (short)0, c,
                                                false, false);
}

// ---- VALU cross-lane reductions (DPP8 within 8 lanes, permlane16 for xor8) -
template <int M>
__device__ __forceinline__ float dpp8_xor_f(float x) {
  constexpr int sel = ((0 ^ M) << 0) | ((1 ^ M) << 3) | ((2 ^ M) << 6) |
                      ((3 ^ M) << 9) | ((4 ^ M) << 12) | ((5 ^ M) << 15) |
                      ((6 ^ M) << 18) | ((7 ^ M) << 21);
  return __int_as_float(__builtin_amdgcn_mov_dpp8(__float_as_int(x), sel));
}
__device__ __forceinline__ float permlane16_xor8_f(float x) {
  int xi = __float_as_int(x);
  // lane i (within each 16-lane row) reads lane i^8
  int r = __builtin_amdgcn_permlane16(xi, xi, 0xFEDCBA98u, 0x76543210u, false,
                                      false);
  return __int_as_float(r);
}
__device__ __forceinline__ float half_row_max(float x) {
  x = fmaxf(x, dpp8_xor_f<1>(x));
  x = fmaxf(x, dpp8_xor_f<2>(x));
  x = fmaxf(x, dpp8_xor_f<4>(x));
  x = fmaxf(x, permlane16_xor8_f(x));
  return x;
}
__device__ __forceinline__ float half_row_sum(float x) {
  x += dpp8_xor_f<1>(x);
  x += dpp8_xor_f<2>(x);
  x += dpp8_xor_f<4>(x);
  x += permlane16_xor8_f(x);
  return x;
}

// ---- Async global -> LDS copy (ASYNCcnt path, CDNA5 ISA) ------------
__device__ __forceinline__ void async_copy_b128(unsigned lds_addr,
                                                const void* gaddr) {
  asm volatile("global_load_async_to_lds_b128 %0, %1, off" ::"v"(lds_addr),
               "v"((unsigned long long)(size_t)gaddr)
               : "memory");
}
__device__ __forceinline__ void wait_async0() {
  asm volatile("s_wait_asynccnt 0x0" ::: "memory");
}
__device__ __forceinline__ unsigned lds_addr_of(const void* p) {
  // generic shared pointer: low 32 bits are the LDS byte offset
  return (unsigned)(size_t)p;
}

// ---- Tensor Data Mover 2D tile load (TENSORcnt path, CDNA5 ISA-8) ---
// D# group0: [1:0]=count=1, [63:32]=lds_addr, [120:64]=global_addr, type=2.
// D# group1: data_size=2B, optional LDS pad (interval/amount codes), tile
// d0 x d1, tensor_dim0_stride in elements.
__device__ __forceinline__ void tdm_load_2d(unsigned lds_addr,
                                            const void* gaddr,
                                            unsigned tile_d0, unsigned tile_d1,
                                            unsigned d0_stride,
                                            unsigned pad_int_code,
                                            unsigned pad_amt_code,
                                            bool pad_en) {
  const unsigned long long ga = (unsigned long long)(size_t)gaddr;
  tdm_u4 g0;
  g0[0] = 1u;                                       // count=1, user mode
  g0[1] = lds_addr;                                 // LDS byte address
  g0[2] = (unsigned)(ga & 0xFFFFFFFFu);             // global_addr[31:0]
  g0[3] = (unsigned)((ga >> 32) & 0x1FFFFFFu) | (2u << 30);  // [56:32] | type=2
  tdm_i8 g1;
  unsigned w0 = (1u << 16);                         // data_size = 2 bytes
  if (pad_en) w0 |= (1u << 20) | (pad_int_code << 22) | (pad_amt_code << 25);
  g1[0] = (int)w0;
  g1[1] = (int)((tile_d0 & 0xFFFFu) << 16);         // tensor_dim0[15:0]
  g1[2] = (int)((tile_d1 & 0xFFFFu) << 16);         // tensor_dim1[15:0]
  g1[3] = (int)((tile_d0 & 0xFFFFu) << 16);         // tile_dim0
  g1[4] = (int)(tile_d1 & 0xFFFFu);                 // tile_dim1 (tile_dim2=0)
  g1[5] = (int)d0_stride;                           // tensor_dim0_stride[31:0]
  g1[6] = 0;
  g1[7] = 0;
  const tdm_i4 gz = {0, 0, 0, 0};
#if defined(__clang_major__) && (__clang_major__ >= 23)
  const tdm_i8 gz8 = {0, 0, 0, 0, 0, 0, 0, 0};
  __builtin_amdgcn_tensor_load_to_lds(g0, g1, gz, gz, gz8, 0);
#else
  __builtin_amdgcn_tensor_load_to_lds(g0, g1, gz, gz, 0);
#endif
}

// ---------------------------------------------------------------------------
// LayerNorm (one row of DIM_=1024 per 256-thread block) -> f16
// ---------------------------------------------------------------------------
__global__ __launch_bounds__(256) void layernorm_f16_kernel(
    const float* __restrict__ x, const float* __restrict__ g,
    const float* __restrict__ bb, _Float16* __restrict__ out) {
  __shared__ float s1[256];
  __shared__ float s2[256];
  const int row = blockIdx.x, t = threadIdx.x;
  const float* xr = x + (size_t)row * DIM_;
  v4f xv = *(const v4f*)(xr + t * 4);
  float sum = xv[0] + xv[1] + xv[2] + xv[3];
  float sq  = xv[0]*xv[0] + xv[1]*xv[1] + xv[2]*xv[2] + xv[3]*xv[3];
  s1[t] = sum; s2[t] = sq;
  __syncthreads();
  for (int off = 128; off > 0; off >>= 1) {
    if (t < off) { s1[t] += s1[t + off]; s2[t] += s2[t + off]; }
    __syncthreads();
  }
  const float mean = s1[0] * (1.0f / DIM_);
  const float var  = s2[0] * (1.0f / DIM_) - mean * mean;
  const float rs   = rsqrtf(var + 1e-5f);
#pragma unroll
  for (int i = 0; i < 4; ++i) {
    int c = t * 4 + i;
    out[(size_t)row * DIM_ + c] = (_Float16)((xv[i] - mean) * rs * g[c] + bb[c]);
  }
}

// ---------------------------------------------------------------------------
// Generic WMMA GEMM:  C(Arows x Ncols) = A_f16(Arows x K) @ W_f32(K x Ncols)
// block = 256 thr (8 waves), tile 64(M) x 128(N), wave tile 32x32, K-step 32.
// A tile staged by one TDM tensor_load_to_lds (wave 0, padded rows 32->40
// halves); W tile transposed+converted in VALU while the DMA runs.
// MODE 0: q scatter (scaled);  MODE 1: k/v scatter;  MODE 2: f32 row-major out
// ---------------------------------------------------------------------------
template <int MODE>
__global__ __launch_bounds__(256) void gemm_wmma_kernel(
    const _Float16* __restrict__ A, const float* __restrict__ W, int K,
    int Ncols, int rowsPerBatch, int posOff, float scale,
    _Float16* __restrict__ out0, _Float16* __restrict__ out1,
    float* __restrict__ outF) {
  __shared__ __align__(16) _Float16 As[64][40];    // 64 rows x 32 K (+pad)
  __shared__ __align__(16) _Float16 Bt[128][40];   // 128 cols x 32 K (+pad)

  const int tid = threadIdx.x, lane = tid & 31, wave = tid >> 5;
  const int wr = wave >> 2, wc = wave & 3;
  const int r = lane & 15, hi = lane >> 4;
  const int row0 = blockIdx.y * 64, n0 = blockIdx.x * 128;

  v8f acc[2][2];
#pragma unroll
  for (int i = 0; i < 2; ++i)
#pragma unroll
    for (int j = 0; j < 2; ++j)
#pragma unroll
      for (int e = 0; e < 8; ++e) acc[i][j][e] = 0.0f;

  for (int k0 = 0; k0 < K; k0 += 32) {
    __syncthreads();
    if (tid < 32) {  // wave 0: TDM pulls the 64x32-half A tile (pad 16->20 dw)
      tdm_load_2d(lds_addr_of(&As[0][0]), A + (size_t)row0 * K + k0,
                  /*tile_d0=*/32, /*tile_d1=*/64, /*stride=*/(unsigned)K,
                  /*pad_interval: 16 dwords*/3, /*pad: 4 dwords*/3, true);
    }
    {  // all waves: stage W tile transposed: 32(k) x 128(n) f32 -> Bt[n][k] f16
      const int kk = tid >> 3, nb = (tid & 7) * 16;
      const v4f* wp = (const v4f*)(W + (size_t)(k0 + kk) * Ncols + n0 + nb);
#pragma unroll
      for (int e = 0; e < 4; ++e) {
        v4f w4 = wp[e];
#pragma unroll
        for (int c = 0; c < 4; ++c) Bt[nb + e * 4 + c][kk] = (_Float16)w4[c];
      }
    }
    if (tid < 32) __builtin_amdgcn_s_wait_tensorcnt(0);
    __syncthreads();

    v16h aF[2], bF[2];
    aF[0] = loadFragA(&As[wr * 32 + r][0], hi);
    aF[1] = loadFragA(&As[wr * 32 + 16 + r][0], hi);
    bF[0] = loadFragB(&Bt[wc * 32 + r][0], hi);
    bF[1] = loadFragB(&Bt[wc * 32 + 16 + r][0], hi);
#pragma unroll
    for (int i = 0; i < 2; ++i)
#pragma unroll
      for (int j = 0; j < 2; ++j) acc[i][j] = wmma_f32(aF[i], bF[j], acc[i][j]);
  }

  // epilogue: C layout -> lane holds col = lane&15, row = reg + 8*hi
#pragma unroll
  for (int i = 0; i < 2; ++i)
#pragma unroll
    for (int j = 0; j < 2; ++j)
#pragma unroll
      for (int reg = 0; reg < 8; ++reg) {
        const int row = row0 + wr * 32 + i * 16 + reg + 8 * hi;
        const int col = n0 + wc * 32 + j * 16 + r;
        const float val = acc[i][j][reg];
        if (MODE == 0) {  // q: (B,H,M,HD), scaled by HD^-0.5
          const int b2 = row / rowsPerBatch, m = row % rowsPerBatch;
          const int h = col >> 6, hd = col & 63;
          out0[(((size_t)(b2 * H_ + h)) * M_ + m) * HD_ + hd] =
              (_Float16)(val * scale);
        } else if (MODE == 1) {  // k / v: (B,H,NM,HD)
          const int b2 = row / rowsPerBatch;
          const int pos = row % rowsPerBatch + posOff;
          const int c2 = col & 1023;
          const int h = c2 >> 6, hd = c2 & 63;
          _Float16* dst = (col >= 1024) ? out1 : out0;
          dst[(((size_t)(b2 * H_ + h)) * NM_ + pos) * HD_ + hd] = (_Float16)val;
        } else {  // MODE 2: plain f32 row-major
          outF[(size_t)row * Ncols + col] = val;
        }
      }
}

// ---------------------------------------------------------------------------
// Rotary on k_x[..., :32] (pairs interleaved), positions 0..N_-1 only.
// ---------------------------------------------------------------------------
__global__ __launch_bounds__(256) void rotary_kernel(
    _Float16* __restrict__ k, const float* __restrict__ cs,
    const float* __restrict__ sn) {
  const int idx = blockIdx.x * 256 + threadIdx.x;  // B*H*N*16 threads
  const int i = idx & 15;
  const int n = (idx >> 4) & (N_ - 1);
  const int bh = idx >> 16;  // / (16*N_)
  _Float16* kp = k + (((size_t)bh * NM_ + n) * HD_) + 2 * i;
  const float x0 = (float)kp[0], x1 = (float)kp[1];
  const float c = cs[n * 16 + i], s = sn[n * 16 + i];
  kp[0] = (_Float16)(x0 * c - x1 * s);
  kp[1] = (_Float16)(x1 * c + x0 * s);
}

// ---------------------------------------------------------------------------
// Flash attention: block = 128 thr (4 waves), one 16-row m-tile per wave.
// grid = B*H*(M/64). K/V streamed in 32-key chunks through LDS; online softmax.
// K chunk staged with async global->LDS; V transposed synchronously.
// ---------------------------------------------------------------------------
__global__ __launch_bounds__(128) void attn_kernel(
    const _Float16* __restrict__ q, const _Float16* __restrict__ k,
    const _Float16* __restrict__ v, _Float16* __restrict__ out) {
  __shared__ __align__(16) _Float16 Ks[32][72];     // 32 keys x 64 d (+pad)
  __shared__ __align__(16) _Float16 Vt[64][40];     // 64 d x 32 keys (+pad)
  __shared__ __align__(16) _Float16 Ps[4][16][40];  // per-wave P transpose

  const int tid = threadIdx.x, wave = tid >> 5, lane = tid & 31;
  const int r = lane & 15, hi = lane >> 4;
  const int mblk = blockIdx.x & 7;   // M_/64 = 8
  const int bh = blockIdx.x >> 3;    // 0..31
  const int b = bh >> 4, h = bh & 15;
  const int m0 = mblk * 64 + wave * 16;

  const _Float16* qrow = q + (((size_t)bh * M_) + m0 + r) * HD_;
  const v16h qf0 = loadFragA(qrow, hi);       // K dims 0..31 (q already scaled)
  const v16h qf1 = loadFragA(qrow + 32, hi);  // K dims 32..63

  const _Float16* kbase = k + (size_t)bh * NM_ * HD_;
  const _Float16* vbase = v + (size_t)bh * NM_ * HD_;

  v8f acc[4];
  float mi[8], li[8];
#pragma unroll
  for (int t = 0; t < 4; ++t)
#pragma unroll
    for (int e = 0; e < 8; ++e) acc[t][e] = 0.0f;
#pragma unroll
  for (int e = 0; e < 8; ++e) { mi[e] = -1.0e30f; li[e] = 0.0f; }

  for (int c0 = 0; c0 < NM_; c0 += 32) {
    __syncthreads();
    {  // stage K chunk via async copy (layout-preserving), V transposed
      const int key = tid >> 2, seg = tid & 3;
      const _Float16* kg = kbase + (size_t)(c0 + key) * HD_ + seg * 16;
      async_copy_b128(lds_addr_of(&Ks[key][seg * 16]), kg);
      async_copy_b128(lds_addr_of(&Ks[key][seg * 16 + 8]), kg + 8);
      const v8h* vg = (const v8h*)(vbase + (size_t)(c0 + key) * HD_ + seg * 16);
      v8h v0 = vg[0], v1 = vg[1];
#pragma unroll
      for (int e = 0; e < 8; ++e) {
        Vt[seg * 16 + e][key] = v0[e];
        Vt[seg * 16 + 8 + e][key] = v1[e];
      }
      if (c0 + 32 < NM_) {  // -> global_prefetch_b8 for the next chunk
        __builtin_prefetch(kbase + (size_t)(c0 + 32 + key) * HD_ + seg * 16, 0, 3);
        __builtin_prefetch(vbase + (size_t)(c0 + 32 + key) * HD_ + seg * 16, 0, 3);
      }
    }
    wait_async0();
    __syncthreads();

    // preload all K fragments, then issue the QK^T WMMA chain
    v16h kf[2][2];
#pragma unroll
    for (int j = 0; j < 2; ++j) {
      kf[j][0] = loadFragB(&Ks[j * 16 + r][0], hi);
      kf[j][1] = loadFragB(&Ks[j * 16 + r][32], hi);
    }
    v8f S[2];
#pragma unroll
    for (int j = 0; j < 2; ++j) {
      v8f sj;
#pragma unroll
      for (int e = 0; e < 8; ++e) sj[e] = 0.0f;
      sj = wmma_f32(qf0, kf[j][0], sj);
      sj = wmma_f32(qf1, kf[j][1], sj);
      S[j] = sj;
    }

    // online softmax: per-row reductions across the 16-lane half-wave (VALU)
#pragma unroll
    for (int reg = 0; reg < 8; ++reg) {
      const float mx = half_row_max(fmaxf(S[0][reg], S[1][reg]));
      const float mnew = fmaxf(mi[reg], mx);
      const float al = __expf(mi[reg] - mnew);
      const float p0 = __expf(S[0][reg] - mnew);
      const float p1 = __expf(S[1][reg] - mnew);
      const float rs = half_row_sum(p0 + p1);
      li[reg] = li[reg] * al + rs;
      mi[reg] = mnew;
#pragma unroll
      for (int t = 0; t < 4; ++t) acc[t][reg] *= al;
      // stash P (C layout: row=reg+8*hi, col=lane&15) for A-layout reload
      const int prow = reg + 8 * hi;
      Ps[wave][prow][r] = (_Float16)p0;
      Ps[wave][prow][16 + r] = (_Float16)p1;
    }

    // P(16x32) @ V(32x64): preload V fragments, then WMMA chain
    const v16h Pf = loadFragA(&Ps[wave][r][0], hi);
    v16h vf[4];
#pragma unroll
    for (int t = 0; t < 4; ++t) vf[t] = loadFragB(&Vt[t * 16 + r][0], hi);
#pragma unroll
    for (int t = 0; t < 4; ++t) acc[t] = wmma_f32(Pf, vf[t], acc[t]);
  }

  // finalize: out[b, m, h*64+d] = acc / l
#pragma unroll
  for (int reg = 0; reg < 8; ++reg) {
    const float inv = 1.0f / li[reg];
    const int m = m0 + reg + 8 * hi;
#pragma unroll
    for (int t = 0; t < 4; ++t) {
      const int d = t * 16 + r;
      out[((size_t)(b * M_ + m)) * INNER_ + h * HD_ + d] =
          (_Float16)(acc[t][reg] * inv);
    }
  }
}

// ---------------------------------------------------------------------------
// Host launcher
// ---------------------------------------------------------------------------
extern "C" void kernel_launch(void* const* d_in, const int* in_sizes, int n_in,
                              void* d_out, int out_size, void* d_ws,
                              size_t ws_size, hipStream_t stream) {
  (void)in_sizes; (void)n_in; (void)out_size; (void)ws_size;
  const float* x    = (const float*)d_in[0];
  const float* lat  = (const float*)d_in[1];
  const float* cs   = (const float*)d_in[2];
  const float* sn   = (const float*)d_in[3];
  const float* Wq   = (const float*)d_in[4];
  const float* Wkv  = (const float*)d_in[5];
  const float* Wout = (const float*)d_in[6];
  const float* gx   = (const float*)d_in[7];
  const float* bx   = (const float*)d_in[8];
  const float* gl   = (const float*)d_in[9];
  const float* bl   = (const float*)d_in[10];
  float* out = (float*)d_out;

  char* ws = (char*)d_ws;
  const size_t sz_xn = (size_t)B_ * N_ * DIM_ * sizeof(_Float16);   // 16 MB
  const size_t sz_ln = (size_t)B_ * M_ * DIM_ * sizeof(_Float16);   //  2 MB
  const size_t sz_q  = (size_t)B_ * H_ * M_ * HD_ * sizeof(_Float16);
  const size_t sz_kv = (size_t)B_ * H_ * NM_ * HD_ * sizeof(_Float16);
  _Float16* xn = (_Float16*)(ws);
  _Float16* lnb = (_Float16*)(ws + sz_xn);
  _Float16* qb = (_Float16*)(ws + sz_xn + sz_ln);
  _Float16* kb = (_Float16*)(ws + sz_xn + sz_ln + sz_q);
  _Float16* vb = (_Float16*)(ws + sz_xn + sz_ln + sz_q + sz_kv);
  _Float16* ao = (_Float16*)(ws + sz_xn + sz_ln + sz_q + 2 * sz_kv);

  // 1) layernorms -> f16
  layernorm_f16_kernel<<<B_ * N_, 256, 0, stream>>>(x, gx, bx, xn);
  layernorm_f16_kernel<<<B_ * M_, 256, 0, stream>>>(lat, gl, bl, lnb);
  // 2) q = ln @ Wq (scale folded in)
  gemm_wmma_kernel<0><<<dim3(INNER_ / 128, (B_ * M_) / 64), 256, 0, stream>>>(
      lnb, Wq, DIM_, INNER_, M_, 0, 0.125f, qb, nullptr, nullptr);
  // 3) k_l, v_l = ln @ Wkv  (positions N..N+M-1)
  gemm_wmma_kernel<1><<<dim3((2 * INNER_) / 128, (B_ * M_) / 64), 256, 0,
                        stream>>>(lnb, Wkv, DIM_, 2 * INNER_, M_, N_, 1.0f, kb,
                                  vb, nullptr);
  // 4) k_x, v_x = xn @ Wkv  (positions 0..N-1) — dominant GEMM
  gemm_wmma_kernel<1><<<dim3((2 * INNER_) / 128, (B_ * N_) / 64), 256, 0,
                        stream>>>(xn, Wkv, DIM_, 2 * INNER_, N_, 0, 1.0f, kb,
                                  vb, nullptr);
  // 5) rotary on k_x[..., :32]
  rotary_kernel<<<(B_ * H_ * N_ * 16) / 256, 256, 0, stream>>>(kb, cs, sn);
  // 6) flash attention -> attn_out f16 (B*M x INNER)
  attn_kernel<<<B_ * H_ * (M_ / 64), 128, 0, stream>>>(qb, kb, vb, ao);
  // 7) final GEMM -> f32 output
  gemm_wmma_kernel<2><<<dim3(DIM_ / 128, (B_ * M_) / 64), 256, 0, stream>>>(
      ao, Wout, INNER_, DIM_, M_, 0, 1.0f, nullptr, nullptr, out);
}
